// VDR_14370960572475
// MI455X (gfx1250) — matrix operand order repaired and checked by reading, not compiled, not run on hardware
//
#include <hip/hip_runtime.h>

// ---------------------------------------------------------------------------
// VDR² (Student-kernel HSIC-style statistic) on MI455X (gfx1250, wave32).
// Strategy: bf16x3-split WMMA Gram tiles, recomputed in two passes so the
// 64MB K matrices never touch HBM; inputs (2MB each) stay L2-resident.
// All reductions are fixed-order -> deterministic across replays.
// ---------------------------------------------------------------------------

typedef __attribute__((ext_vector_type(16))) __bf16 v16bf;
typedef __attribute__((ext_vector_type(8)))  float  v8f;

#define NROW    4096
#define KDIM    128
#define NSTRIPS (NROW / 16)            // 256 row-strips of 16 rows
#define WAVES_PER_BLOCK 8              // 256 threads
#define TILES_PER_WAVE  (NSTRIPS / WAVES_PER_BLOCK)  // 32 column tiles per wave

// GAMMA = gamma(0.8) / (gamma(0.3) * sqrt(0.6)) ; exponent -(sigma+1)/2 = -0.8
#define GAMMA_C 0.5024166f
#define NEXP    (-0.8f)
#define INV_N   (1.0f / 4096.0f)

union ABFrag { uint4 u[2]; v16bf v; };   // 32 bytes = 16 bf16 = one WMMA fragment/lane

__device__ __forceinline__ unsigned short f2bf_rne(float f) {
  unsigned u = __float_as_uint(f);
  unsigned r = u + 0x7FFFu + ((u >> 16) & 1u);   // round-to-nearest-even
  return (unsigned short)(r >> 16);
}

__device__ __forceinline__ float student_k(float G, float asq, float bsq) {
  float d2   = asq + bsq - 2.0f * G;
  float base = fmaxf(1.0f + 0.5f * d2, 1e-30f);
  // base^-0.8 = exp2(-0.8 * log2(base)); v_log_f32 / v_exp_f32 hardware trans ops
  return GAMMA_C * __builtin_amdgcn_exp2f(NEXP * __builtin_amdgcn_logf(base));
}

// ---- A-matrix fragments (16x32 bf16 per K-chunk), ISA 7.12.2 layout --------
// lane<16: row m=lane, K = {kb..kb+7, kb+16..kb+23}
// lane>=16: row m=lane-16, K = {kb+8..kb+15, kb+24..kb+31}
__device__ __forceinline__ void load_a(const unsigned short* __restrict__ Ah,
                                       const unsigned short* __restrict__ Al,
                                       int rowBase, int mrow, int half,
                                       ABFrag ah[4], ABFrag al[4]) {
  const uint4* Ah4 = (const uint4*)(Ah + (size_t)(rowBase + mrow) * KDIM);
  const uint4* Al4 = (const uint4*)(Al + (size_t)(rowBase + mrow) * KDIM);
#pragma unroll
  for (int kb4 = 0; kb4 < 4; ++kb4) {            // kb = 32*kb4
    ah[kb4].u[0] = Ah4[kb4 * 4 + half];          // elems kb + 8*half
    ah[kb4].u[1] = Ah4[kb4 * 4 + 2 + half];      // elems kb + 16 + 8*half
    al[kb4].u[0] = Al4[kb4 * 4 + half];
    al[kb4].u[1] = Al4[kb4 * 4 + 2 + half];
  }
}

// ---- one 16x16 Gram tile, K=128, bf16x3 split (hi*hi + hi*lo + lo*hi) ------
// B layout (32x16 bf16): lane n=lane&15 holds K = {kb+8*half..+7, kb+16+8*half..+7}.
__device__ __forceinline__ v8f gram_tile(const unsigned short* __restrict__ Bh,
                                         const unsigned short* __restrict__ Bl,
                                         int colBase, int mrow, int half,
                                         const ABFrag ah[4], const ABFrag al[4]) {
  v8f acc = {};
  const uint4* Bh4 = (const uint4*)(Bh + (size_t)(colBase + mrow) * KDIM);
  const uint4* Bl4 = (const uint4*)(Bl + (size_t)(colBase + mrow) * KDIM);
#pragma unroll
  for (int kb4 = 0; kb4 < 4; ++kb4) {
    ABFrag bh, bl;
    bh.u[0] = Bh4[kb4 * 4 + half];
    bh.u[1] = Bh4[kb4 * 4 + 2 + half];
    bl.u[0] = Bl4[kb4 * 4 + half];
    bl.u[1] = Bl4[kb4 * 4 + 2 + half];
    acc = __builtin_amdgcn_wmma_f32_16x16x32_bf16(false, ah[kb4].v, false, bh.v,
                                                  (short)0, acc, false, false);
    acc = __builtin_amdgcn_wmma_f32_16x16x32_bf16(false, ah[kb4].v, false, bl.v,
                                                  (short)0, acc, false, false);
    acc = __builtin_amdgcn_wmma_f32_16x16x32_bf16(false, al[kb4].v, false, bh.v,
                                                  (short)0, acc, false, false);
  }
  return acc;
}

// ---------------------------------------------------------------------------
// Kernel 1: bf16 hi/lo split of X1/X2 + fp32 squared row norms.
// grid (4096, 2), block 128 (one row per block).
// ---------------------------------------------------------------------------
__global__ void __launch_bounds__(128)
prep_kernel(const float* __restrict__ X1, const float* __restrict__ X2,
            unsigned short* __restrict__ X1h, unsigned short* __restrict__ X1l,
            unsigned short* __restrict__ X2h, unsigned short* __restrict__ X2l,
            float* __restrict__ xsq, float* __restrict__ ysq) {
  const int row = blockIdx.x;
  const int t   = threadIdx.x;
  const float* src; unsigned short* dh; unsigned short* dl; float* sq;
  if (blockIdx.y == 0) { src = X1; dh = X1h; dl = X1l; sq = xsq; }
  else                 { src = X2; dh = X2h; dl = X2l; sq = ysq; }

  float f = src[(size_t)row * KDIM + t];
  unsigned short h = f2bf_rne(f);
  float hf = __uint_as_float((unsigned)h << 16);
  unsigned short l = f2bf_rne(f - hf);
  dh[(size_t)row * KDIM + t] = h;
  dl[(size_t)row * KDIM + t] = l;

  float s = f * f;
#pragma unroll
  for (int m = 1; m < 32; m <<= 1) s += __shfl_xor(s, m, 32);
  __shared__ float ls[4];
  if ((t & 31) == 0) ls[t >> 5] = s;
  __syncthreads();
  if (t == 0) sq[row] = ((ls[0] + ls[1]) + ls[2]) + ls[3];
}

// ---------------------------------------------------------------------------
// Kernel 2: per-row sums of K over full 4096 columns.
// grid (256, 4): y = 0:XX  1:YY  2:XY (rows of X1)  3:YX (rows of X2 -> XY colsums)
// block 256 = 8 waves, strip of 16 rows; each wave owns 32 contiguous 16-col tiles.
// ---------------------------------------------------------------------------
__global__ void __launch_bounds__(256)
rowsum_kernel(const unsigned short* __restrict__ X1h, const unsigned short* __restrict__ X1l,
              const unsigned short* __restrict__ X2h, const unsigned short* __restrict__ X2l,
              const float* __restrict__ xsq, const float* __restrict__ ysq,
              float* __restrict__ rsXX, float* __restrict__ rsYY,
              float* __restrict__ rsXY, float* __restrict__ rsYX) {
  const unsigned short *Ah, *Al, *Bh, *Bl; const float *asq, *bsq; float* out;
  switch (blockIdx.y) {
    case 0:  Ah = X1h; Al = X1l; Bh = X1h; Bl = X1l; asq = xsq; bsq = xsq; out = rsXX; break;
    case 1:  Ah = X2h; Al = X2l; Bh = X2h; Bl = X2l; asq = ysq; bsq = ysq; out = rsYY; break;
    case 2:  Ah = X1h; Al = X1l; Bh = X2h; Bl = X2l; asq = xsq; bsq = ysq; out = rsXY; break;
    default: Ah = X2h; Al = X2l; Bh = X1h; Bl = X1l; asq = ysq; bsq = xsq; out = rsYX; break;
  }
  const int rowBase = blockIdx.x * 16;
  const int lane = threadIdx.x & 31, wave = threadIdx.x >> 5;
  const int half = lane >> 4, mrow = lane & 15;

  ABFrag ah[4], al[4];
  load_a(Ah, Al, rowBase, mrow, half, ah, al);
  float asqv[8];
#pragma unroll
  for (int v = 0; v < 8; ++v) asqv[v] = asq[rowBase + v + 8 * half];

  float rowacc[8];
#pragma unroll
  for (int v = 0; v < 8; ++v) rowacc[v] = 0.0f;

  for (int jt = 0; jt < TILES_PER_WAVE; ++jt) {
    const int colBase = (wave * TILES_PER_WAVE + jt) * 16;
    if (jt + 1 < TILES_PER_WAVE)
      __builtin_prefetch(Bh + (size_t)(colBase + 16 + mrow) * KDIM, 0, 1);
    v8f acc = gram_tile(Bh, Bl, colBase, mrow, half, ah, al);
    float bs = bsq[colBase + mrow];
#pragma unroll
    for (int v = 0; v < 8; ++v)
      rowacc[v] += student_k(acc[v], asqv[v], bs);
  }

  // reduce across the 16 lanes of each half (rows are per-(v,half))
#pragma unroll
  for (int v = 0; v < 8; ++v) {
    float r = rowacc[v];
    r += __shfl_xor(r, 1, 32);
    r += __shfl_xor(r, 2, 32);
    r += __shfl_xor(r, 4, 32);
    r += __shfl_xor(r, 8, 32);
    rowacc[v] = r;
  }
  __shared__ float ls[WAVES_PER_BLOCK][16];
  if (lane == 0)  { for (int v = 0; v < 8; ++v) ls[wave][v]     = rowacc[v]; }
  if (lane == 16) { for (int v = 0; v < 8; ++v) ls[wave][8 + v] = rowacc[v]; }
  __syncthreads();
  if (threadIdx.x < 16) {
    float r = 0.0f;
#pragma unroll
    for (int w = 0; w < WAVES_PER_BLOCK; ++w) r += ls[w][threadIdx.x];
    out[rowBase + threadIdx.x] = r;
  }
}

// ---------------------------------------------------------------------------
// Kernel 3: totals[mat] = sum of rowsums (fixed-order tree). grid 3, block 128.
// ---------------------------------------------------------------------------
__global__ void __launch_bounds__(128)
totals_kernel(const float* __restrict__ rsXX, const float* __restrict__ rsYY,
              const float* __restrict__ rsXY, float* __restrict__ totals) {
  const float* rs = (blockIdx.x == 0) ? rsXX : (blockIdx.x == 1) ? rsYY : rsXY;
  const int t = threadIdx.x;
  float s = 0.0f;
  for (int i = t; i < NROW; i += 128) s += rs[i];
  __shared__ float ls[128];
  ls[t] = s; __syncthreads();
  for (int st = 64; st > 0; st >>= 1) { if (t < st) ls[t] += ls[t + st]; __syncthreads(); }
  if (t == 0) totals[blockIdx.x] = ls[0];
}

// ---------------------------------------------------------------------------
// Kernel 4: recompute K tiles, double-center, accumulate squared values.
// grid (256, 3): y = 0:XX 1:YY 2:XY. One fp32 partial per block.
// ---------------------------------------------------------------------------
__global__ void __launch_bounds__(256)
csq_kernel(const unsigned short* __restrict__ X1h, const unsigned short* __restrict__ X1l,
           const unsigned short* __restrict__ X2h, const unsigned short* __restrict__ X2l,
           const float* __restrict__ xsq, const float* __restrict__ ysq,
           const float* __restrict__ rsXX, const float* __restrict__ rsYY,
           const float* __restrict__ rsXY, const float* __restrict__ rsYX,
           const float* __restrict__ totals, float* __restrict__ partial2) {
  const unsigned short *Ah, *Al, *Bh, *Bl; const float *asq, *bsq, *rs, *cs;
  switch (blockIdx.y) {
    case 0:  Ah = X1h; Al = X1l; Bh = X1h; Bl = X1l; asq = xsq; bsq = xsq; rs = rsXX; cs = rsXX; break;
    case 1:  Ah = X2h; Al = X2l; Bh = X2h; Bl = X2l; asq = ysq; bsq = ysq; rs = rsYY; cs = rsYY; break;
    default: Ah = X1h; Al = X1l; Bh = X2h; Bl = X2l; asq = xsq; bsq = ysq; rs = rsXY; cs = rsYX; break;
  }
  const float tmean = totals[blockIdx.y] * (INV_N * INV_N);
  const int rowBase = blockIdx.x * 16;
  const int lane = threadIdx.x & 31, wave = threadIdx.x >> 5;
  const int half = lane >> 4, mrow = lane & 15;

  ABFrag ah[4], al[4];
  load_a(Ah, Al, rowBase, mrow, half, ah, al);
  float asqv[8], rmv[8];
#pragma unroll
  for (int v = 0; v < 8; ++v) {
    asqv[v] = asq[rowBase + v + 8 * half];
    rmv[v]  = rs[rowBase + v + 8 * half] * INV_N;
  }

  float sacc = 0.0f;
  for (int jt = 0; jt < TILES_PER_WAVE; ++jt) {
    const int colBase = (wave * TILES_PER_WAVE + jt) * 16;
    if (jt + 1 < TILES_PER_WAVE)
      __builtin_prefetch(Bh + (size_t)(colBase + 16 + mrow) * KDIM, 0, 1);
    v8f acc = gram_tile(Bh, Bl, colBase, mrow, half, ah, al);
    float bs = bsq[colBase + mrow];
    float cm = cs[colBase + mrow] * INV_N;
#pragma unroll
    for (int v = 0; v < 8; ++v) {
      float Kv = student_k(acc[v], asqv[v], bs);
      float c  = Kv - rmv[v] - cm + tmean;
      sacc = fmaf(c, c, sacc);
    }
  }

  sacc += __shfl_xor(sacc, 1, 32);
  sacc += __shfl_xor(sacc, 2, 32);
  sacc += __shfl_xor(sacc, 4, 32);
  sacc += __shfl_xor(sacc, 8, 32);
  sacc += __shfl_xor(sacc, 16, 32);
  __shared__ float ls[WAVES_PER_BLOCK];
  if (lane == 0) ls[wave] = sacc;
  __syncthreads();
  if (threadIdx.x == 0) {
    float r = 0.0f;
#pragma unroll
    for (int w = 0; w < WAVES_PER_BLOCK; ++w) r += ls[w];
    partial2[blockIdx.y * NSTRIPS + blockIdx.x] = r;
  }
}

// ---------------------------------------------------------------------------
// Kernel 5: final fixed-order reduce + unbiased combination -> scalar.
// ---------------------------------------------------------------------------
__global__ void __launch_bounds__(256)
final_kernel(const float* __restrict__ partial2, float* __restrict__ out) {
  __shared__ float s0[256], s1[256], s2[256];
  const int t = threadIdx.x;
  s0[t] = partial2[t];
  s1[t] = partial2[256 + t];
  s2[t] = partial2[512 + t];
  __syncthreads();
  for (int st = 128; st > 0; st >>= 1) {
    if (t < st) { s0[t] += s0[t + st]; s1[t] += s1[t + st]; s2[t] += s2[t + st]; }
    __syncthreads();
  }
  if (t == 0) {
    const float mf = 4096.0f;
    float vdr2 = (s0[0] - mf) / ((mf - 1.0f) * mf)
               + (s1[0] - mf) / ((mf - 1.0f) * mf)
               - 2.0f * s2[0] / (mf * mf);
    out[0] = vdr2;
  }
}

// ---------------------------------------------------------------------------
extern "C" void kernel_launch(void* const* d_in, const int* in_sizes, int n_in,
                              void* d_out, int out_size, void* d_ws, size_t ws_size,
                              hipStream_t stream) {
  (void)in_sizes; (void)n_in; (void)out_size; (void)ws_size;
  const float* X1 = (const float*)d_in[0];
  const float* X2 = (const float*)d_in[1];
  float* out = (float*)d_out;

  // workspace layout (~4.2 MB), 256B aligned sections
  char* ws = (char*)d_ws;
  size_t off = 0;
  auto take = [&](size_t bytes) -> void* {
    void* p = ws + off;
    off += (bytes + 255) & ~(size_t)255;
    return p;
  };
  unsigned short* X1h = (unsigned short*)take((size_t)NROW * KDIM * 2);
  unsigned short* X1l = (unsigned short*)take((size_t)NROW * KDIM * 2);
  unsigned short* X2h = (unsigned short*)take((size_t)NROW * KDIM * 2);
  unsigned short* X2l = (unsigned short*)take((size_t)NROW * KDIM * 2);
  float* xsq   = (float*)take(NROW * 4);
  float* ysq   = (float*)take(NROW * 4);
  float* rsXX  = (float*)take(NROW * 4);
  float* rsYY  = (float*)take(NROW * 4);
  float* rsXY  = (float*)take(NROW * 4);
  float* rsYX  = (float*)take(NROW * 4);
  float* totals   = (float*)take(4 * 4);
  float* partial2 = (float*)take(3 * NSTRIPS * 4);

  prep_kernel<<<dim3(NROW, 2), 128, 0, stream>>>(X1, X2, X1h, X1l, X2h, X2l, xsq, ysq);
  rowsum_kernel<<<dim3(NSTRIPS, 4), 256, 0, stream>>>(X1h, X1l, X2h, X2l, xsq, ysq,
                                                      rsXX, rsYY, rsXY, rsYX);
  totals_kernel<<<dim3(3), 128, 0, stream>>>(rsXX, rsYY, rsXY, totals);
  csq_kernel<<<dim3(NSTRIPS, 3), 256, 0, stream>>>(X1h, X1l, X2h, X2l, xsq, ysq,
                                                   rsXX, rsYY, rsXY, rsYX, totals, partial2);
  final_kernel<<<dim3(1), 256, 0, stream>>>(partial2, out);
}
